// RWKVTimeMix_35699768164399
// MI455X (gfx1250) — matrix operand order repaired and checked by reading, not compiled, not run on hardware
//
#include <hip/hip_runtime.h>
#include <hip/hip_bf16.h>

// RWKV TimeMix for MI455X (gfx1250, wave32).
// - GEMMs: bf16x3 split-precision WMMA (v_wmma_f32_16x16x32_bf16), f32 accum.
// - B (weight) tiles staged to LDS by the Tensor Data Mover (tensor_load_to_lds),
//   double-buffered, with TDM-inserted padding for conflict-free LDS reads.
// - WKV scan: chunk-parallel 3-pass log-sum-exp scan (32 chunks of 128 steps).

#define Bdim 8
#define Tdim 4096
#define Cdim 1024
#define Mdim (Bdim * Tdim)   // 32768

#define NCH 32               // chunks per sequence
#define CHL (Tdim / NCH)     // 128 steps per chunk

typedef __attribute__((ext_vector_type(16))) __bf16 v16bf;
typedef __attribute__((ext_vector_type(8)))  float  v8f;
typedef __attribute__((ext_vector_type(4)))  unsigned int v4u;
typedef __attribute__((ext_vector_type(8)))  int v8i;
typedef __attribute__((ext_vector_type(4)))  int v4i;

union FragAB {
  v16bf v;
  uint4 q[2];
};

#if __has_builtin(__builtin_amdgcn_tensor_load_to_lds)
#define USE_TDM 1
#else
#define USE_TDM 0
#endif

// ---------------------------------------------------------------------------
// Split f32 weight matrix into bf16 hi + bf16 lo (residual) arrays.
// ---------------------------------------------------------------------------
__global__ void wsplit_kernel(const float* __restrict__ W,
                              __bf16* __restrict__ hi,
                              __bf16* __restrict__ lo, int n) {
  int i = blockIdx.x * blockDim.x + threadIdx.x;
  if (i < n) {
    float f = W[i];
    __bf16 h = (__bf16)f;
    hi[i] = h;
    lo[i] = (__bf16)(f - (float)h);
  }
}

// ---------------------------------------------------------------------------
// TDM: async-load a 2D bf16 tile [128 rows x 32 cols] from W (row stride Cdim)
// into LDS. Pad 4 dwords after every 16 dwords -> 80B LDS row stride
// (conflict-free for the 16B fragment reads: 20-bank step, gcd(20,64)=4).
// D# packing per CDNA5 ISA ch.8 (groups 0/1; 2D tensor so groups 2/3 zero).
// This toolchain exposes the 6-arg builtin form (extra int32x8 before cpol).
// ---------------------------------------------------------------------------
#if USE_TDM
__device__ __forceinline__ void tdm_load_tile_bf16(unsigned ldsOff,
                                                   const __bf16* gptr) {
  unsigned long long ga = (unsigned long long)(uintptr_t)gptr;
  v4u g0;
  g0[0] = 1u;                                   // count=1, no gather
  g0[1] = ldsOff;                               // lds_addr (bytes)
  g0[2] = (unsigned)(ga & 0xFFFFFFFFu);         // global_addr[31:0]
  g0[3] = (unsigned)((ga >> 32) & 0x1FFFFFFu)   // global_addr[56:32]
          | (2u << 30);                         // type = 2 ("image")
  v8i g1;
  g1[0] = (int)((1u << 16)      // data_size = 2 bytes
              | (1u << 20)      // pad_enable
              | (3u << 22)      // pad_interval: 16 dwords (64B)
              | (3u << 25));    // pad_amount: 4 dwords (16B)
  g1[1] = (int)(1024u << 16);   // tensor_dim0 = 1024 (elems)
  g1[2] = (int)(1024u << 16);   // tensor_dim0 hi=0 | tensor_dim1 = 1024 lo
  g1[3] = (int)(32u << 16);     // tensor_dim1 hi=0 | tile_dim0 = 32
  g1[4] = 128;                  // tile_dim1 = 128, tile_dim2 = 0
  g1[5] = 1024;                 // tensor_dim0_stride = 1024 (elems)
  g1[6] = 0;
  g1[7] = 0;
  v4i z4 = {0, 0, 0, 0};
  v8i z8 = {0, 0, 0, 0, 0, 0, 0, 0};
  __builtin_amdgcn_tensor_load_to_lds(g0, g1, z4, z4, z8, 0);
}
#endif

// ---------------------------------------------------------------------------
// out[m][n] = sum_c A'[m][c] * W[n][c]   (A' = lerp(x_prev, x, mix) if doShift)
// BLK 64(M) x 128(N), 8 waves of 32, each wave = 32x32 (2x2 WMMA tiles),
// K-step 32, bf16x3: acc += Ahi*Bhi + Ahi*Blo + Alo*Bhi.
// ---------------------------------------------------------------------------
#define BLK_M 64
#define BLK_N 128
#define LDA   40   // bf16 elems per LDS row (80B) -> conflict-free b128 reads

__global__ __launch_bounds__(256)
void gemm_bf16x3_kernel(const float* __restrict__ A,     // [Mdim, Cdim] f32
                        const float* __restrict__ mix,   // [Cdim] or nullptr
                        const __bf16* __restrict__ Whi,  // [Cdim, Cdim]
                        const __bf16* __restrict__ Wlo,  // [Cdim, Cdim]
                        float* __restrict__ out,         // [Mdim, Cdim] f32
                        int doSigmoid, int doShift) {
  __shared__ __bf16 AsHi[BLK_M * LDA];
  __shared__ __bf16 AsLo[BLK_M * LDA];
#if USE_TDM
  __shared__ __bf16 BsHi[2][BLK_N * LDA];   // TDM-written, 80B row stride
  __shared__ __bf16 BsLo[2][BLK_N * LDA];
#endif

  const int tid   = threadIdx.x;
  const int wave  = tid >> 5;
  const int lane  = tid & 31;
  const int lhalf = lane >> 4;   // 0 or 1
  const int l15   = lane & 15;

  const int blockN = blockIdx.x * BLK_N;
  const int blockM = blockIdx.y * BLK_M;

  const int wm = (wave & 1) * 32;   // wave row offset inside block
  const int wn = (wave >> 1) * 32;  // wave col offset inside block

  v8f acc[2][2];
#pragma unroll
  for (int i = 0; i < 2; ++i)
#pragma unroll
    for (int j = 0; j < 2; ++j)
#pragma unroll
      for (int r = 0; r < 8; ++r) acc[i][j][r] = 0.0f;

  // A-tile loader mapping: 4 threads per row, 8 consecutive K each.
  const int lrow = tid >> 2;        // 0..63
  const int lk   = (tid & 3) * 8;   // 0,8,16,24
  const int gm_l = blockM + lrow;
  const int tmod = gm_l & (Tdim - 1);  // position within sequence

#if USE_TDM
  // Prologue: kick off DMA of the first B tiles (wave 0 issues; EXEC ignored).
  if (wave == 0) {
    tdm_load_tile_bf16((unsigned)(uintptr_t)&BsHi[0][0],
                       Whi + (size_t)blockN * Cdim + 0);
    tdm_load_tile_bf16((unsigned)(uintptr_t)&BsLo[0][0],
                       Wlo + (size_t)blockN * Cdim + 0);
  }
  int cur = 0;
#endif

  for (int it = 0; it < Cdim / 32; ++it) {
    const int k0 = it * 32;
    // ---- stage A tile (fused time-shift lerp + bf16 hi/lo split) ----
    {
      const float* xrow = A + (size_t)gm_l * Cdim + (k0 + lk);
      float xv[8];
      *(float4*)&xv[0] = *(const float4*)xrow;
      *(float4*)&xv[4] = *(const float4*)(xrow + 4);
      if (doShift) {
        float xp[8];
        if (tmod != 0) {
          const float* prow = xrow - Cdim;  // previous token, same batch
          *(float4*)&xp[0] = *(const float4*)prow;
          *(float4*)&xp[4] = *(const float4*)(prow + 4);
        } else {
#pragma unroll
          for (int e = 0; e < 8; ++e) xp[e] = 0.0f;
        }
        float mx[8];
        *(float4*)&mx[0] = *(const float4*)(mix + k0 + lk);
        *(float4*)&mx[4] = *(const float4*)(mix + k0 + lk + 4);
#pragma unroll
        for (int e = 0; e < 8; ++e) xv[e] = xp[e] + mx[e] * (xv[e] - xp[e]);
      }
#pragma unroll
      for (int e = 0; e < 8; ++e) {
        float f = xv[e];
        __bf16 h = (__bf16)f;
        AsHi[lrow * LDA + lk + e] = h;
        AsLo[lrow * LDA + lk + e] = (__bf16)(f - (float)h);
      }
    }
#if USE_TDM
    __builtin_amdgcn_s_wait_tensorcnt(0);  // B tiles for this iter landed
#endif
    __syncthreads();
#if USE_TDM
    // Overlap: DMA next K-step's B tiles while this step computes.
    if (wave == 0 && it + 1 < Cdim / 32) {
      const int kn = k0 + 32;
      tdm_load_tile_bf16((unsigned)(uintptr_t)&BsHi[cur ^ 1][0],
                         Whi + (size_t)blockN * Cdim + kn);
      tdm_load_tile_bf16((unsigned)(uintptr_t)&BsLo[cur ^ 1][0],
                         Wlo + (size_t)blockN * Cdim + kn);
    }
#endif

    // ---- load fragments ----
    // A frag (16-bit 16x32): lane = row (m = lane&15), K = 16*half + e.
    // Two 16B LDS reads per frag.
    FragAB ahi[2], alo[2], bhi[2], blo[2];
#pragma unroll
    for (int i = 0; i < 2; ++i) {
      const int row = wm + i * 16 + l15;
      ahi[i].q[0] = *(const uint4*)&AsHi[row * LDA + 16 * lhalf];
      ahi[i].q[1] = *(const uint4*)&AsHi[row * LDA + 16 * lhalf + 8];
      alo[i].q[0] = *(const uint4*)&AsLo[row * LDA + 16 * lhalf];
      alo[i].q[1] = *(const uint4*)&AsLo[row * LDA + 16 * lhalf + 8];
    }
    // B frag (16-bit 32x16): lane = col (n = lane&15), K = 16*half + e.
#pragma unroll
    for (int j = 0; j < 2; ++j) {
      const int nl = wn + j * 16 + l15;
#if USE_TDM
      const __bf16* bh = &BsHi[cur][nl * LDA + 16 * lhalf];
      bhi[j].q[0] = *(const uint4*)bh;
      bhi[j].q[1] = *(const uint4*)(bh + 8);
      const __bf16* bl = &BsLo[cur][nl * LDA + 16 * lhalf];
      blo[j].q[0] = *(const uint4*)bl;
      blo[j].q[1] = *(const uint4*)(bl + 8);
#else
      const int n = blockN + nl;
      const __bf16* wp = Whi + (size_t)n * Cdim + k0 + 16 * lhalf;
      bhi[j].q[0] = *(const uint4*)wp;
      bhi[j].q[1] = *(const uint4*)(wp + 8);
      const __bf16* wq = Wlo + (size_t)n * Cdim + k0 + 16 * lhalf;
      blo[j].q[0] = *(const uint4*)wq;
      blo[j].q[1] = *(const uint4*)(wq + 8);
#endif
    }

    // ---- bf16x3 WMMA ----
#pragma unroll
    for (int i = 0; i < 2; ++i)
#pragma unroll
      for (int j = 0; j < 2; ++j) {
        acc[i][j] = __builtin_amdgcn_wmma_f32_16x16x32_bf16(
            false, ahi[i].v, false, bhi[j].v, (short)0, acc[i][j], false, false);
        acc[i][j] = __builtin_amdgcn_wmma_f32_16x16x32_bf16(
            false, ahi[i].v, false, blo[j].v, (short)0, acc[i][j], false, false);
        acc[i][j] = __builtin_amdgcn_wmma_f32_16x16x32_bf16(
            false, alo[i].v, false, bhi[j].v, (short)0, acc[i][j], false, false);
      }
    __syncthreads();
#if USE_TDM
    cur ^= 1;
#endif
  }

  // ---- epilogue: C/D layout: lane -> N = lane&15, vgpr r -> M = r + 8*half ----
#pragma unroll
  for (int i = 0; i < 2; ++i) {
    const int gmBase = blockM + wm + i * 16 + 8 * lhalf;
#pragma unroll
    for (int j = 0; j < 2; ++j) {
      const int gn = blockN + wn + j * 16 + l15;
#pragma unroll
      for (int r = 0; r < 8; ++r) {
        float v = acc[i][j][r];
        if (doSigmoid) v = 1.0f / (1.0f + expf(-v));
        out[(size_t)(gmBase + r) * Cdim + gn] = v;
      }
    }
  }
}

// ---------------------------------------------------------------------------
// Chunk-parallel WKV scan. Un-stabilized recurrence is linear:
//   A_t = e^w A_{t-1} + e^{k_t} v_t ,  B_t likewise
// so running L steps on incoming state S equals decay(S, L*w) merged with the
// chunk-local state computed from zero. States kept stabilized as (a,b,p)
// meaning (e^p a, e^p b).
// ---------------------------------------------------------------------------
__global__ void wkv_pass1_kernel(const float* __restrict__ td,
                                 const float* __restrict__ kbuf,
                                 const float* __restrict__ vbuf,
                                 float* __restrict__ sA, float* __restrict__ sB,
                                 float* __restrict__ sP) {
  const int i = blockIdx.x * blockDim.x + threadIdx.x;  // B*NCH*C threads
  const int c = i & (Cdim - 1);
  const int j = (i >> 10) & (NCH - 1);
  const int b = i >> 15;
  const float w = -expf(td[c]);

  float a = 0.0f, bb = 0.0f, p = -1e38f;
  size_t idx = ((size_t)b * Tdim + (size_t)j * CHL) * Cdim + c;
  for (int t = 0; t < CHL; ++t, idx += Cdim) {
    const float kt = kbuf[idx];
    const float vt = vbuf[idx];
    const float q2 = fmaxf(p + w, kt);
    const float f1 = expf(p + w - q2);
    const float f2 = expf(kt - q2);
    a = f1 * a + f2 * vt;
    bb = f1 * bb + f2;
    p = q2;
  }
  const size_t s = (size_t)((b * NCH + j) << 10) + c;
  sA[s] = a; sB[s] = bb; sP[s] = p;
}

__global__ void wkv_pass2_kernel(const float* __restrict__ td,
                                 const float* __restrict__ sA,
                                 const float* __restrict__ sB,
                                 const float* __restrict__ sP,
                                 float* __restrict__ iA, float* __restrict__ iB,
                                 float* __restrict__ iP) {
  const int i = blockIdx.x * blockDim.x + threadIdx.x;  // B*C threads
  const int c = i & (Cdim - 1);
  const int b = i >> 10;
  const float w = -expf(td[c]);
  const float Lw = (float)CHL * w;

  float a = 0.0f, bb = 0.0f, p = -1e38f;
  for (int j = 0; j < NCH; ++j) {
    const size_t s = (size_t)((b * NCH + j) << 10) + c;
    iA[s] = a; iB[s] = bb; iP[s] = p;       // incoming state of chunk j
    const float pd = p + Lw;                 // decay incoming over chunk j
    const float cp = sP[s];
    const float q = fmaxf(pd, cp);
    const float e1 = expf(pd - q);
    const float e2 = expf(cp - q);
    a  = e1 * a  + e2 * sA[s];
    bb = e1 * bb + e2 * sB[s];
    p = q;
  }
}

__global__ void wkv_pass3_kernel(const float* __restrict__ td,
                                 const float* __restrict__ tf,
                                 float* __restrict__ kbuf,        // in: k, out: sr*wkv
                                 const float* __restrict__ vbuf,
                                 const float* __restrict__ srbuf,
                                 const float* __restrict__ iA,
                                 const float* __restrict__ iB,
                                 const float* __restrict__ iP) {
  const int i = blockIdx.x * blockDim.x + threadIdx.x;  // B*NCH*C threads
  const int c = i & (Cdim - 1);
  const int j = (i >> 10) & (NCH - 1);
  const int b = i >> 15;
  const float w = -expf(td[c]);
  const float u = tf[c];

  const size_t s = (size_t)((b * NCH + j) << 10) + c;
  float a = iA[s], bb = iB[s], p = iP[s];

  size_t idx = ((size_t)b * Tdim + (size_t)j * CHL) * Cdim + c;
  for (int t = 0; t < CHL; ++t, idx += Cdim) {
    const float kt = kbuf[idx];
    const float vt = vbuf[idx];
    // output (with bonus u on current token)
    const float q  = fmaxf(p, u + kt);
    const float e1 = expf(p - q);
    const float e2 = expf(u + kt - q);
    const float y  = (e1 * a + e2 * vt) / (e1 * bb + e2);
    kbuf[idx] = y * srbuf[idx];
    // state update with decay w
    const float q2 = fmaxf(p + w, kt);
    const float f1 = expf(p + w - q2);
    const float f2 = expf(kt - q2);
    a = f1 * a + f2 * vt;
    bb = f1 * bb + f2;
    p = q2;
  }
}

// ---------------------------------------------------------------------------
extern "C" void kernel_launch(void* const* d_in, const int* in_sizes, int n_in,
                              void* d_out, int out_size, void* d_ws, size_t ws_size,
                              hipStream_t stream) {
  const float* x  = (const float*)d_in[0];
  const float* td = (const float*)d_in[1];
  const float* tf = (const float*)d_in[2];
  const float* mk = (const float*)d_in[3];
  const float* mv = (const float*)d_in[4];
  const float* mr = (const float*)d_in[5];
  const float* Wk = (const float*)d_in[6];
  const float* Wv = (const float*)d_in[7];
  const float* Wr = (const float*)d_in[8];
  const float* Wo = (const float*)d_in[9];
  float* out = (float*)d_out;

  // Workspace layout (~406 MB):
  //   8 x bf16[C*C]       weight hi/lo splits       (16 MB)
  //   3 x f32[B*T*C]      k, v, sr buffers          (384 MB)
  //   6 x f32[B*NCH*C]    scan chunk/incoming state (6 MB)
  char* ws = (char*)d_ws;
  const size_t wmat = (size_t)Cdim * Cdim;
  __bf16* WhiK = (__bf16*)ws;
  __bf16* WloK = WhiK + wmat;
  __bf16* WhiV = WloK + wmat;
  __bf16* WloV = WhiV + wmat;
  __bf16* WhiR = WloV + wmat;
  __bf16* WloR = WhiR + wmat;
  __bf16* WhiO = WloR + wmat;
  __bf16* WloO = WhiO + wmat;
  float* kbuf  = (float*)(ws + 8 * wmat * sizeof(__bf16));
  float* vbuf  = kbuf + (size_t)Mdim * Cdim;
  float* srbuf = vbuf + (size_t)Mdim * Cdim;
  float* sA    = srbuf + (size_t)Mdim * Cdim;
  const size_t nst = (size_t)Bdim * NCH * Cdim;
  float* sB = sA + nst;
  float* sP = sB + nst;
  float* iA = sP + nst;
  float* iB = iA + nst;
  float* iP = iB + nst;

  // 1) split weights to bf16 hi/lo
  const int nW = (int)wmat;
  const int wsBlocks = (nW + 255) / 256;
  wsplit_kernel<<<wsBlocks, 256, 0, stream>>>(Wk, WhiK, WloK, nW);
  wsplit_kernel<<<wsBlocks, 256, 0, stream>>>(Wv, WhiV, WloV, nW);
  wsplit_kernel<<<wsBlocks, 256, 0, stream>>>(Wr, WhiR, WloR, nW);
  wsplit_kernel<<<wsBlocks, 256, 0, stream>>>(Wo, WhiO, WloO, nW);

  // 2) k, v, sigmoid(r) GEMMs with fused time-shift lerp
  dim3 grid(Cdim / BLK_N, Mdim / BLK_M);
  gemm_bf16x3_kernel<<<grid, 256, 0, stream>>>(x, mk, WhiK, WloK, kbuf, 0, 1);
  gemm_bf16x3_kernel<<<grid, 256, 0, stream>>>(x, mv, WhiV, WloV, vbuf, 0, 1);
  gemm_bf16x3_kernel<<<grid, 256, 0, stream>>>(x, mr, WhiR, WloR, srbuf, 1, 1);

  // 3) chunk-parallel WKV scan (writes sr*wkv in place over kbuf)
  const int scanThreads = Bdim * NCH * Cdim;  // 262144
  wkv_pass1_kernel<<<scanThreads / 256, 256, 0, stream>>>(td, kbuf, vbuf, sA, sB, sP);
  wkv_pass2_kernel<<<(Bdim * Cdim) / 256, 256, 0, stream>>>(td, sA, sB, sP, iA, iB, iP);
  wkv_pass3_kernel<<<scanThreads / 256, 256, 0, stream>>>(td, tf, kbuf, vbuf, srbuf,
                                                          iA, iB, iP);

  // 4) output projection
  gemm_bf16x3_kernel<<<grid, 256, 0, stream>>>(kbuf, nullptr, WhiO, WloO, out, 0, 0);
}
